// VectorQuantizer_47777216200711
// MI455X (gfx1250) — matrix operand order repaired and checked by reading, not compiled, not run on hardware
//
#include <hip/hip_runtime.h>
#include <hip/hip_bf16.h>

// ---------------- problem constants ----------------
#define BATCH   64
#define DIMS    256
#define HWSZ    1024            // 32*32
#define NTOK    65536           // BATCH*HWSZ
#define NCODES  1024
#define BETA    0.25f

// d_out layout: z_q_st (NTOK*DIMS), loss (1), indices (NTOK)
#define OUT_LOSS_OFF  ((size_t)NTOK * DIMS)
#define OUT_IDX_OFF   ((size_t)NTOK * DIMS + 1)

// workspace layout (bytes)
#define WS_ZHI     ((size_t)0)
#define WS_ZLO     ((size_t)(NTOK) * DIMS * 2)                    // 33554432
#define WS_CHI     (WS_ZLO + (size_t)(NTOK) * DIMS * 2)           // 67108864
#define WS_CLO     (WS_CHI + (size_t)NCODES * DIMS * 2)           // +524288
#define WS_CNORM   (WS_CLO + (size_t)NCODES * DIMS * 2)           // +524288
#define WS_IDX     (WS_CNORM + (size_t)NCODES * 4)
#define WS_PARTIAL (WS_IDX + (size_t)NTOK * 4)
#define NPARTIAL   65536                                          // gather-kernel block count

// codebook LDS staging
#define CHUNK      32                      // codes per LDS chunk (= 2 WMMA tiles)
#define NCHUNK     (NCODES / CHUNK)        // 32 chunks
#define LDS_ROW    264                     // padded ushorts per code row (bank stride 4)
#define CHUNK_USH  (CHUNK * LDS_ROW)       // 8448 ushorts = 16896 B per array

typedef __attribute__((ext_vector_type(16))) __bf16          v16bf;
typedef __attribute__((ext_vector_type(8)))  float           v8f;
typedef __attribute__((ext_vector_type(8)))  unsigned short  us8;
typedef __attribute__((ext_vector_type(16))) unsigned short  us16;

__device__ __forceinline__ unsigned short f32_to_bf16(float f) {
    unsigned int u = __float_as_uint(f);
    u += 0x7FFFu + ((u >> 16) & 1u);           // round-to-nearest-even
    return (unsigned short)(u >> 16);
}
__device__ __forceinline__ float bf16_to_f32(unsigned short h) {
    return __uint_as_float(((unsigned int)h) << 16);
}
__device__ __forceinline__ v16bf cat16(us8 lo, us8 hi) {
    us16 w;
#pragma unroll
    for (int i = 0; i < 8; ++i) { w[i] = lo[i]; w[i + 8] = hi[i]; }
    return __builtin_bit_cast(v16bf, w);
}

// Async DMA one 32x256 bf16 chunk (row-major, rows padded to LDS_ROW in LDS).
// 32 rows * 32 segments of 16B = 1024 b128 transfers; 256 threads -> 4 each.
__device__ __forceinline__ void async_copy_chunk(const unsigned short* __restrict__ gbase,
                                                 unsigned ldsbase, int tid) {
#pragma unroll
    for (int i = 0; i < 4; ++i) {
        int s = i * 256 + tid;
        int row = s >> 5;
        int col = (s & 31) * 8;
        unsigned goff = (unsigned)((row * DIMS + col) * 2);
        unsigned loff = ldsbase + (unsigned)((row * LDS_ROW + col) * 2);
        asm volatile("global_load_async_to_lds_b128 %0, %1, %2"
                     :: "v"(loff), "v"(goff), "s"(gbase) : "memory");
    }
}
__device__ __forceinline__ void wait_async_zero() {
    asm volatile("s_wait_asynccnt 0x0" ::: "memory");
}

// ---------------- kernel 1: codebook norms + bf16 hi/lo split ----------------
__global__ __launch_bounds__(256) void vq_prep_codebook(
    const float* __restrict__ cb, unsigned short* __restrict__ chi,
    unsigned short* __restrict__ clo, float* __restrict__ cnorm) {
    int j = blockIdx.x, t = threadIdx.x;
    float c = cb[(size_t)j * DIMS + t];
    unsigned short h = f32_to_bf16(c);
    chi[(size_t)j * DIMS + t] = h;
    clo[(size_t)j * DIMS + t] = f32_to_bf16(c - bf16_to_f32(h));
    __shared__ float red[256];
    red[t] = c * c;
    __syncthreads();
    for (int s = 128; s > 0; s >>= 1) {
        if (t < s) red[t] += red[t + s];
        __syncthreads();
    }
    if (t == 0) cnorm[j] = red[0];
}

// ---------------- kernel 2: (B,D,H,W) f32 -> (N,D) bf16 hi/lo (LDS transpose) ----------------
__global__ __launch_bounds__(256) void vq_convert_z(
    const float* __restrict__ z, unsigned short* __restrict__ zhi,
    unsigned short* __restrict__ zlo) {
    __shared__ float tile[32][33];
    int b = blockIdx.z, hw0 = blockIdx.x * 32, d0 = blockIdx.y * 32;
    const float* zb = z + (size_t)b * DIMS * HWSZ;
    int tx = threadIdx.x, ty = threadIdx.y;
#pragma unroll
    for (int i = 0; i < 4; ++i) {
        int dl = ty + 8 * i;
        tile[dl][tx] = zb[(size_t)(d0 + dl) * HWSZ + hw0 + tx];   // coalesced over hw
    }
    __syncthreads();
#pragma unroll
    for (int i = 0; i < 4; ++i) {
        int hwl = ty + 8 * i;
        float v = tile[tx][hwl];
        unsigned short h = f32_to_bf16(v);
        size_t o = (size_t)(b * HWSZ + hw0 + hwl) * DIMS + d0 + tx; // coalesced over d
        zhi[o] = h;
        zlo[o] = f32_to_bf16(v - bf16_to_f32(h));
    }
}

// ---------------- kernel 3: split-bf16 WMMA distance GEMM + fused argmin ----------------
// One wave per 16-token tile, 8 waves/block. Codebook double-buffered in LDS via
// async global->LDS DMA; all 8 waves share each staged chunk. The two 16-code
// tiles of a chunk are computed concurrently (acc0/acc1) so consecutive WMMAs
// are independent (no RAW hazard slots) and ds_load latency hides behind WMMAs.
__global__ __launch_bounds__(256) void vq_gemm_argmin(
    const unsigned short* __restrict__ zhi, const unsigned short* __restrict__ zlo,
    const unsigned short* __restrict__ chi, const unsigned short* __restrict__ clo,
    const float* __restrict__ cnorm, int* __restrict__ idxout) {
    __shared__ unsigned short sb[2][2][CHUNK_USH];   // [buffer][hi/lo][padded chunk]

    const int tid  = threadIdx.x;
    const int lane = tid & 31;
    const int wave = tid >> 5;
    const int rowBase = (blockIdx.x * 8 + wave) * 16;
    const int half = lane >> 4;        // 0: K-low / rows 0-7 side, 1: K-high / rows 8-15
    const int ln   = lane & 15;

    // ---- preload full A (16 rows x 256 K), hi+lo fragments, reused for all code tiles ----
    const unsigned short* zah = zhi + (size_t)(rowBase + ln) * DIMS;
    const unsigned short* zal = zlo + (size_t)(rowBase + ln) * DIMS;
    v16bf Ahi[8], Alo[8];
#pragma unroll
    for (int kc = 0; kc < 8; ++kc) {
        int k0 = kc * 32 + half * 8;   // A layout: lane<16 -> K [k0,k0+8)+[k0+16,k0+24)
        Ahi[kc] = cat16(*(const us8*)(zah + k0), *(const us8*)(zah + k0 + 16));
        Alo[kc] = cat16(*(const us8*)(zal + k0), *(const us8*)(zal + k0 + 16));
    }

    float bestv[8];
    int   besti[8];
#pragma unroll
    for (int r = 0; r < 8; ++r) { bestv[r] = 3.4e38f; besti[r] = 0; }

    // ---- prologue: stage chunk 0 ----
    async_copy_chunk(chi, (unsigned)(uintptr_t)&sb[0][0][0], tid);
    async_copy_chunk(clo, (unsigned)(uintptr_t)&sb[0][1][0], tid);
    wait_async_zero();
    __syncthreads();

    for (int c = 0; c < NCHUNK; ++c) {
        const int cur = c & 1;
        // prefetch next chunk into spare buffer (overlaps the WMMA phase below)
        if (c + 1 < NCHUNK) {
            const int nb = (c + 1) & 1;
            const size_t goff = (size_t)(c + 1) * CHUNK * DIMS;
            async_copy_chunk(chi + goff, (unsigned)(uintptr_t)&sb[nb][0][0], tid);
            async_copy_chunk(clo + goff, (unsigned)(uintptr_t)&sb[nb][1][0], tid);
        }

        const float cn0 = cnorm[c * CHUNK + ln];
        const float cn1 = cnorm[c * CHUNK + 16 + ln];
        // B layout: lane n holds codebook row (tile*16 + n&15), K [half*8,+8)+[half*8+16,+8)
        const unsigned short* cbh0 = &sb[cur][0][0] + ln * LDS_ROW + half * 8;
        const unsigned short* cbl0 = &sb[cur][1][0] + ln * LDS_ROW + half * 8;
        const unsigned short* cbh1 = cbh0 + 16 * LDS_ROW;
        const unsigned short* cbl1 = cbl0 + 16 * LDS_ROW;

        v8f acc0 = {};
        v8f acc1 = {};
#pragma unroll
        for (int kc = 0; kc < 8; ++kc) {
            v16bf Bhi0 = cat16(*(const us8*)(cbh0 + kc * 32), *(const us8*)(cbh0 + kc * 32 + 16));
            v16bf Blo0 = cat16(*(const us8*)(cbl0 + kc * 32), *(const us8*)(cbl0 + kc * 32 + 16));
            v16bf Bhi1 = cat16(*(const us8*)(cbh1 + kc * 32), *(const us8*)(cbh1 + kc * 32 + 16));
            v16bf Blo1 = cat16(*(const us8*)(cbl1 + kc * 32), *(const us8*)(cbl1 + kc * 32 + 16));
            // (zh+zl)(ch+cl) ~= zh*ch + zl*ch + zh*cl ; interleave the two tiles
            acc0 = __builtin_amdgcn_wmma_f32_16x16x32_bf16(false, Ahi[kc], false, Bhi0, (short)0, acc0, false, false);
            acc1 = __builtin_amdgcn_wmma_f32_16x16x32_bf16(false, Ahi[kc], false, Bhi1, (short)0, acc1, false, false);
            acc0 = __builtin_amdgcn_wmma_f32_16x16x32_bf16(false, Alo[kc], false, Bhi0, (short)0, acc0, false, false);
            acc1 = __builtin_amdgcn_wmma_f32_16x16x32_bf16(false, Alo[kc], false, Bhi1, (short)0, acc1, false, false);
            acc0 = __builtin_amdgcn_wmma_f32_16x16x32_bf16(false, Ahi[kc], false, Blo0, (short)0, acc0, false, false);
            acc1 = __builtin_amdgcn_wmma_f32_16x16x32_bf16(false, Ahi[kc], false, Blo1, (short)0, acc1, false, false);
        }

#pragma unroll
        for (int tt = 0; tt < 2; ++tt) {
            const v8f& acc = tt ? acc1 : acc0;
            const float cn = tt ? cn1 : cn0;
            const int code = c * CHUNK + tt * 16 + ln;
#pragma unroll
            for (int r = 0; r < 8; ++r) {
                // distance (mod. row const): ||c||^2 - 2 z.c ; row = r + 8*half, col = ln
                float v = cn - 2.0f * acc[r];
                int id = code;
#pragma unroll
                for (int off = 1; off < 16; off <<= 1) {   // min across 16 lanes of each half
                    float ov = __shfl_xor(v, off);
                    int   oi = __shfl_xor(id, off);
                    if (ov < v || (ov == v && oi < id)) { v = ov; id = oi; }
                }
                if (v < bestv[r] || (v == bestv[r] && id < besti[r])) { bestv[r] = v; besti[r] = id; }
            }
        }
        wait_async_zero();     // next chunk's DMA has landed
        __syncthreads();       // everyone done reading cur, next buffer valid
    }

    if (ln == 0) {
#pragma unroll
        for (int r = 0; r < 8; ++r) idxout[rowBase + half * 8 + r] = besti[r];
    }
}

// ---------------- kernel 4: gather z_q, write outputs, per-block loss partials ----------------
__global__ __launch_bounds__(256) void vq_gather_out(
    const float* __restrict__ z, const float* __restrict__ cb,
    const int* __restrict__ idx, float* __restrict__ out,
    float* __restrict__ partial) {
    int b = blockIdx.z;
    int hw = blockIdx.x * 32 + threadIdx.x;
    int d  = blockIdx.y * 8 + threadIdx.y;
    int n  = b * HWSZ + hw;
    int code = idx[n];
    float q = cb[(size_t)code * DIMS + d];
    size_t o = (size_t)b * DIMS * HWSZ + (size_t)d * HWSZ + hw;
    float diff = z[o] - q;
    out[o] = q;                                   // z_q_st forward value == z_q
    __shared__ float red[256];
    int t = threadIdx.y * 32 + threadIdx.x;
    red[t] = diff * diff;
    __syncthreads();
    for (int s = 128; s > 0; s >>= 1) {
        if (t < s) red[t] += red[t + s];
        __syncthreads();
    }
    int blk = (blockIdx.z * gridDim.y + blockIdx.y) * gridDim.x + blockIdx.x;
    if (t == 0) partial[blk] = red[0];
    if (blockIdx.y == 0 && threadIdx.y == 0)      // indices as float, once per token
        out[OUT_IDX_OFF + n] = (float)code;
}

// ---------------- kernel 5: deterministic loss reduction ----------------
__global__ __launch_bounds__(256) void vq_loss_reduce(
    const float* __restrict__ partial, float* __restrict__ out) {
    __shared__ float red[256];
    int t = threadIdx.x;
    float s = 0.0f;
    for (int i = t; i < NPARTIAL; i += 256) s += partial[i];
    red[t] = s;
    __syncthreads();
    for (int k = 128; k > 0; k >>= 1) {
        if (t < k) red[t] += red[t + k];
        __syncthreads();
    }
    if (t == 0) out[OUT_LOSS_OFF] = red[0] * (BETA / (float)((size_t)NTOK * DIMS));
}

extern "C" void kernel_launch(void* const* d_in, const int* in_sizes, int n_in,
                              void* d_out, int out_size, void* d_ws, size_t ws_size,
                              hipStream_t stream) {
    const float* z  = (const float*)d_in[0];   // (64,256,32,32) f32
    const float* cb = (const float*)d_in[1];   // (1024,256) f32
    float* out = (float*)d_out;
    char*  ws  = (char*)d_ws;

    unsigned short* zhi = (unsigned short*)(ws + WS_ZHI);
    unsigned short* zlo = (unsigned short*)(ws + WS_ZLO);
    unsigned short* chi = (unsigned short*)(ws + WS_CHI);
    unsigned short* clo = (unsigned short*)(ws + WS_CLO);
    float* cnorm        = (float*)(ws + WS_CNORM);
    int*   idx          = (int*)(ws + WS_IDX);
    float* partial      = (float*)(ws + WS_PARTIAL);

    vq_prep_codebook<<<NCODES, 256, 0, stream>>>(cb, chi, clo, cnorm);
    vq_convert_z<<<dim3(HWSZ / 32, DIMS / 32, BATCH), dim3(32, 8), 0, stream>>>(z, zhi, zlo);
    vq_gemm_argmin<<<NTOK / 128, 256, 0, stream>>>(zhi, zlo, chi, clo, cnorm, idx);
    vq_gather_out<<<dim3(HWSZ / 32, DIMS / 8, BATCH), dim3(32, 8), 0, stream>>>(z, cb, idx, out, partial);
    vq_loss_reduce<<<1, 256, 0, stream>>>(partial, out);
}
